// PointNet2SurfaceRegressor_83425444757844
// MI455X (gfx1250) — compile-verified, matrix operand order checked by reading, not run on hardware
//
#include <hip/hip_runtime.h>
#include <hip/hip_bf16.h>
#include <cstdint>
#include <cstddef>

typedef __attribute__((ext_vector_type(16))) _Float16 v16h;
typedef __attribute__((ext_vector_type(8)))  float    v8f;

#define BBATCH 32
#define NPTS   1225
#define KNN    32
#define MAXPTS 1232
#define MAXS2  640
#define LDH    40   // padded LDS row stride in halfs (80B: keeps 16B chunk align)

union AccU  { v8f v; float f[8]; };
union PKU   { _Float16 h[2]; unsigned u; };
union FragU { v16h v; uint4 q[2]; };

// ---------------------------------------------------------------------------
// GEMM helpers: stage A (64x32) / B (32x128 transposed) tiles into LDS as f16.
// Template bools select bounds checking; steady-state instantiation is
// branch-free straight-line code.
// ---------------------------------------------------------------------------
template <bool CHKM, bool CHKK>
__device__ __forceinline__ void stageA(const float* __restrict__ W,
                                       _Float16 (*As)[LDH],
                                       int tid, int m0, int kk, int M, int K)
{
    #pragma unroll
    for (int i = 0; i < 4; ++i) {
        int idx = tid + (i << 8);          // 0..1023
        int r   = idx >> 4;                 // 0..63
        int c2  = (idx & 15) << 1;          // 0..30
        int gm  = m0 + r;
        int gk  = kk + c2;
        float f0, f1;
        if (!CHKM && !CHKK) {
            const float* s = W + (size_t)gm * K + gk;
            f0 = s[0]; f1 = s[1];
        } else {
            bool okm = !CHKM || (gm < M);
            f0 = (okm && (!CHKK || gk     < K)) ? W[(size_t)gm * K + gk]     : 0.f;
            f1 = (okm && (!CHKK || gk + 1 < K)) ? W[(size_t)gm * K + gk + 1] : 0.f;
        }
        PKU pk; pk.h[0] = (_Float16)f0; pk.h[1] = (_Float16)f1;
        *(unsigned*)&As[r][c2] = pk.u;
    }
}

template <bool CHKP, bool CHKK, bool PF>
__device__ __forceinline__ void stageB(const float* __restrict__ X,
                                       _Float16 (*Bs)[LDH],
                                       int tid, int p0, int kk, int K, int P)
{
    #pragma unroll
    for (int i = 0; i < 8; ++i) {
        int idx = tid + (i << 8);          // 0..2047
        int p   = idx & 127;
        int k2  = (idx >> 7) << 1;          // 0,2,..,30
        int gp  = p0 + p;
        int gk  = kk + k2;
        float f0, f1;
        if (!CHKP && !CHKK) {
            const float* s = X + (size_t)gk * P + gp;
            f0 = s[0]; f1 = s[P];
            if (PF && gk + 32 < K)
                __builtin_prefetch(s + (size_t)32 * P, 0, 1);
        } else {
            bool okp = !CHKP || (gp < P);
            f0 = (okp && (!CHKK || gk     < K)) ? X[(size_t)gk * P + gp]       : 0.f;
            f1 = (okp && (!CHKK || gk + 1 < K)) ? X[(size_t)(gk + 1) * P + gp] : 0.f;
        }
        PKU pk; pk.h[0] = (_Float16)f0; pk.h[1] = (_Float16)f1;
        *(unsigned*)&Bs[p][k2] = pk.u;
    }
}

__device__ __forceinline__ void wmmaStep(const _Float16 (*As)[LDH],
                                         const _Float16 (*Bs)[LDH],
                                         int wm, int wn, int mrow, int khalf,
                                         AccU* acc)
{
    FragU a;
    const int arow = (wm << 4) + mrow;
    a.q[0] = *(const uint4*)&As[arow][khalf];
    a.q[1] = *(const uint4*)&As[arow][16 + khalf];
    #pragma unroll
    for (int t = 0; t < 4; ++t) {
        FragU b;
        const int brow = (wn << 6) + (t << 4) + mrow;
        b.q[0] = *(const uint4*)&Bs[brow][khalf];
        b.q[1] = *(const uint4*)&Bs[brow][16 + khalf];
        acc[t].v = __builtin_amdgcn_wmma_f32_16x16x32_f16(
            false, a.v, false, b.v, (short)0, acc[t].v, false, false);
    }
}

// ---------------------------------------------------------------------------
// Generic GEMM:  Y[M][P] = W[M][K] * X[K][P]   (f32 in/out, f16 WMMA compute)
// block = 256 threads = 8 waves; tile 64(M) x 128(P).
// wave w: rows (w&3)*16, cols (w>>2)*64 (four 16-wide WMMA tiles).
// LDS: A row-major [64][32->LDH]; B stored transposed [128][32->LDH] so both
// fragments are two contiguous 16B chunks per lane -> ds_load_b128.
// ---------------------------------------------------------------------------
__global__ __launch_bounds__(256) void k_gemm(const float* __restrict__ W,
                                              const float* __restrict__ X,
                                              float* __restrict__ Y,
                                              int M, int K, int P,
                                              const float* __restrict__ bias,
                                              int relu)
{
    __shared__ __align__(16) _Float16 As[64][LDH];
    __shared__ __align__(16) _Float16 Bs[128][LDH];

    const int tid   = threadIdx.x;
    const int lane  = tid & 31;           // wave32
    const int wave  = tid >> 5;
    const int wm    = wave & 3;           // 16-row strip
    const int wn    = wave >> 2;          // 64-col strip (0..1)
    const int mrow  = lane & 15;
    const int khalf = (lane >> 4) << 3;   // 0 or 8
    const int m0    = blockIdx.y * 64;
    const int p0    = blockIdx.x * 128;

    AccU acc[4];
    #pragma unroll
    for (int t = 0; t < 4; ++t)
        #pragma unroll
        for (int i = 0; i < 8; ++i) acc[t].f[i] = 0.f;

    const bool fullM = (m0 + 64 <= M);
    const bool fullP = (p0 + 128 <= P);
    const int  kfull = K & ~31;

    if (fullM && fullP) {
        for (int kk = 0; kk < kfull; kk += 32) {
            stageA<false, false>(W, As, tid, m0, kk, M, K);
            stageB<false, false, true>(X, Bs, tid, p0, kk, K, P);
            __syncthreads();
            wmmaStep(As, Bs, wm, wn, mrow, khalf, acc);
            __syncthreads();
        }
    } else {
        for (int kk = 0; kk < kfull; kk += 32) {
            stageA<true, false>(W, As, tid, m0, kk, M, K);
            stageB<true, false, false>(X, Bs, tid, p0, kk, K, P);
            __syncthreads();
            wmmaStep(As, Bs, wm, wn, mrow, khalf, acc);
            __syncthreads();
        }
    }
    if (kfull < K) {
        stageA<true, true>(W, As, tid, m0, kfull, M, K);
        stageB<true, true, false>(X, Bs, tid, p0, kfull, K, P);
        __syncthreads();
        wmmaStep(As, Bs, wm, wn, mrow, khalf, acc);
        __syncthreads();
    }

    // C/D layout: VGPR r -> M = r (lanes 0-15) or r+8 (lanes 16-31); N = lane&15
    #pragma unroll
    for (int r = 0; r < 8; ++r) {
        int m = m0 + (wm << 4) + r + khalf;
        if (m >= M) continue;
        float add = bias ? bias[m] : 0.f;
        #pragma unroll
        for (int t = 0; t < 4; ++t) {
            int n = p0 + (wn << 6) + (t << 4) + mrow;
            if (n < P) {
                float v = acc[t].f[r] + add;
                if (relu) v = fmaxf(v, 0.f);
                Y[(size_t)m * P + n] = v;
            }
        }
    }
}

// ---------------------------------------------------------------------------
// BatchNorm (training mode, biased variance) helpers
// ---------------------------------------------------------------------------
__global__ __launch_bounds__(256) void k_bnstats(const float* __restrict__ X,
                                                 int P, float* __restrict__ stats)
{
    int m = blockIdx.x;
    const float* row = X + (size_t)m * P;
    float s = 0.f, q = 0.f;
    for (int i = threadIdx.x; i < P; i += 256) {
        float v = row[i]; s += v; q += v * v;
    }
    __shared__ float rs[256], rq[256];
    rs[threadIdx.x] = s; rq[threadIdx.x] = q;
    __syncthreads();
    for (int o = 128; o > 0; o >>= 1) {
        if (threadIdx.x < o) {
            rs[threadIdx.x] += rs[threadIdx.x + o];
            rq[threadIdx.x] += rq[threadIdx.x + o];
        }
        __syncthreads();
    }
    if (threadIdx.x == 0) {
        float mean = rs[0] / (float)P;
        float var  = rq[0] / (float)P - mean * mean;
        stats[2 * m]     = mean;
        stats[2 * m + 1] = rsqrtf(fmaxf(var, 0.f) + 1e-5f);
    }
}

__global__ void k_bnapply(float* __restrict__ X, const float* __restrict__ stats,
                          const float* __restrict__ g, const float* __restrict__ bb,
                          int P, long long total, int relu)
{
    long long t = (long long)blockIdx.x * blockDim.x + threadIdx.x;
    if (t >= total) return;
    int m = (int)(t / P);
    float v = X[t];
    v = (v - stats[2 * m]) * stats[2 * m + 1] * g[m] + bb[m];
    if (relu) v = fmaxf(v, 0.f);
    X[t] = v;
}

// ---------------------------------------------------------------------------
// Layout helpers
// ---------------------------------------------------------------------------
__global__ void k_xyz_cmaj(const float* __restrict__ xyz, float* __restrict__ out, int BN)
{
    int t = blockIdx.x * blockDim.x + threadIdx.x;
    if (t >= BN) return;
    out[t]          = xyz[t * 3 + 0];
    out[BN + t]     = xyz[t * 3 + 1];
    out[2 * BN + t] = xyz[t * 3 + 2];
}

// ---------------------------------------------------------------------------
// Farthest point sampling: one block per batch, dist array + points in LDS
// ---------------------------------------------------------------------------
__global__ __launch_bounds__(256) void k_fps(const float* __restrict__ xyz,
                                             int N, int S, int* __restrict__ fidx)
{
    int b = blockIdx.x;
    __shared__ float sx[MAXPTS], sy[MAXPTS], sz[MAXPTS], sd[MAXPTS];
    __shared__ float rv[256];
    __shared__ int   ri[256];
    __shared__ int   sfar;
    const float* base = xyz + (size_t)b * N * 3;
    for (int i = threadIdx.x; i < N; i += 256) {
        sx[i] = base[i * 3 + 0];
        sy[i] = base[i * 3 + 1];
        sz[i] = base[i * 3 + 2];
        sd[i] = 1e10f;
    }
    if (threadIdx.x == 0) sfar = 0;
    __syncthreads();
    for (int s = 0; s < S; ++s) {
        int far = sfar;
        if (threadIdx.x == 0) fidx[b * S + s] = far;
        float cx = sx[far], cy = sy[far], cz = sz[far];
        float bestv = -1.f; int besti = 0;
        for (int i = threadIdx.x; i < N; i += 256) {
            float dx = sx[i] - cx, dy = sy[i] - cy, dz = sz[i] - cz;
            float d  = dx * dx + dy * dy + dz * dz;
            float nd = fminf(sd[i], d);
            sd[i] = nd;
            if (nd > bestv) { bestv = nd; besti = i; }
        }
        rv[threadIdx.x] = bestv; ri[threadIdx.x] = besti;
        __syncthreads();
        for (int o = 128; o > 0; o >>= 1) {
            if (threadIdx.x < o) {
                float v2 = rv[threadIdx.x + o]; int i2 = ri[threadIdx.x + o];
                if (v2 > rv[threadIdx.x] ||
                    (v2 == rv[threadIdx.x] && i2 < ri[threadIdx.x])) {
                    rv[threadIdx.x] = v2; ri[threadIdx.x] = i2;
                }
            }
            __syncthreads();
        }
        if (threadIdx.x == 0) sfar = ri[0];
        __syncthreads();
    }
}

__global__ void k_gather_xyz(const float* __restrict__ xyz, const int* __restrict__ fidx,
                             float* __restrict__ nxyz, int Nprev, int S, int BS)
{
    int t = blockIdx.x * blockDim.x + threadIdx.x;
    if (t >= BS) return;
    int b = t / S;
    int j = fidx[t];
    const float* p = xyz + ((size_t)b * Nprev + j) * 3;
    nxyz[t * 3 + 0] = p[0];
    nxyz[t * 3 + 1] = p[1];
    nxyz[t * 3 + 2] = p[2];
}

// ---------------------------------------------------------------------------
// KNN (k = 32 smallest, ties by lower index, matching stable argsort)
// ---------------------------------------------------------------------------
__global__ __launch_bounds__(128) void k_knn(const float* __restrict__ xyz,
                                             const float* __restrict__ nxyz,
                                             int N, int S, int* __restrict__ kidx)
{
    __shared__ float sx[MAXPTS], sy[MAXPTS], sz[MAXPTS];
    int b = blockIdx.y;
    const float* base = xyz + (size_t)b * N * 3;
    for (int i = threadIdx.x; i < N; i += blockDim.x) {
        sx[i] = base[i * 3 + 0];
        sy[i] = base[i * 3 + 1];
        sz[i] = base[i * 3 + 2];
    }
    __syncthreads();
    int s = blockIdx.x * blockDim.x + threadIdx.x;
    if (s >= S) return;
    const float* q = nxyz + ((size_t)b * S + s) * 3;
    float qx = q[0], qy = q[1], qz = q[2];
    float bd[KNN]; int bi[KNN];
    #pragma unroll
    for (int i = 0; i < KNN; ++i) { bd[i] = 3.0e38f; bi[i] = 0; }
    for (int n = 0; n < N; ++n) {
        float dx = sx[n] - qx, dy = sy[n] - qy, dz = sz[n] - qz;
        float d  = dx * dx + dy * dy + dz * dz;
        if (d < bd[KNN - 1]) {
            int j = KNN - 1;
            while (j > 0 && bd[j - 1] > d) {
                bd[j] = bd[j - 1]; bi[j] = bi[j - 1]; --j;
            }
            bd[j] = d; bi[j] = n;
        }
    }
    int* o = kidx + ((size_t)b * S + s) * KNN;
    for (int i = 0; i < KNN; ++i) o[i] = bi[i];
}

// ---------------------------------------------------------------------------
// Grouping: out[(3+Cf)][Q] ; channels 0..2 = relative xyz, 3.. = gathered feats
// q = (b*S + s)*KNN + k
// ---------------------------------------------------------------------------
__global__ void k_group(const float* __restrict__ xyz, const float* __restrict__ nxyz,
                        const float* __restrict__ feats, const int* __restrict__ kidx,
                        float* __restrict__ out, int Nprev, int S, int Cf,
                        int BNprev, long long Q)
{
    long long q = (long long)blockIdx.x * blockDim.x + threadIdx.x;
    if (q >= Q) return;
    long long bs = q >> 5;
    int b = (int)(bs / S);
    int j = kidx[q];
    const float* pj = xyz + ((size_t)b * Nprev + j) * 3;
    const float* pc = nxyz + (size_t)bs * 3;
    out[q]                 = pj[0] - pc[0];
    out[(size_t)Q + q]     = pj[1] - pc[1];
    out[(size_t)2 * Q + q] = pj[2] - pc[2];
    size_t fb = (size_t)b * Nprev + j;
    for (int c = 0; c < Cf; ++c)
        out[(size_t)(3 + c) * Q + q] = feats[(size_t)c * BNprev + fb];
}

__global__ void k_maxk(const float* __restrict__ X, float* __restrict__ out,
                       int C, int BS)
{
    long long t = (long long)blockIdx.x * blockDim.x + threadIdx.x;
    if (t >= (long long)C * BS) return;
    int c = (int)(t / BS);
    int p = (int)(t % BS);
    const float* src = X + ((size_t)c * BS + p) * KNN;
    float m = src[0];
    for (int k = 1; k < KNN; ++k) m = fmaxf(m, src[k]);
    out[t] = m;
}

// ---------------------------------------------------------------------------
// 3-NN + inverse-distance weights
// ---------------------------------------------------------------------------
__global__ __launch_bounds__(128) void k_3nn(const float* __restrict__ xyz1,
                                             const float* __restrict__ xyz2,
                                             int N1, int S2,
                                             int* __restrict__ nidx,
                                             float* __restrict__ nw)
{
    __shared__ float sx[MAXS2], sy[MAXS2], sz[MAXS2];
    int b = blockIdx.y;
    const float* base = xyz2 + (size_t)b * S2 * 3;
    for (int i = threadIdx.x; i < S2; i += blockDim.x) {
        sx[i] = base[i * 3 + 0];
        sy[i] = base[i * 3 + 1];
        sz[i] = base[i * 3 + 2];
    }
    __syncthreads();
    int n = blockIdx.x * blockDim.x + threadIdx.x;
    if (n >= N1) return;
    const float* q = xyz1 + ((size_t)b * N1 + n) * 3;
    float qx = q[0], qy = q[1], qz = q[2];
    float d0 = 3e38f, d1 = 3e38f, d2 = 3e38f;
    int   i0 = 0, i1 = 0, i2 = 0;
    for (int j = 0; j < S2; ++j) {
        float dx = sx[j] - qx, dy = sy[j] - qy, dz = sz[j] - qz;
        float d  = dx * dx + dy * dy + dz * dz;
        if (d < d0)      { d2 = d1; i2 = i1; d1 = d0; i1 = i0; d0 = d; i0 = j; }
        else if (d < d1) { d2 = d1; i2 = i1; d1 = d;  i1 = j; }
        else if (d < d2) { d2 = d;  i2 = j; }
    }
    float r0 = 1.f / (fmaxf(d0, 0.f) + 1e-8f);
    float r1 = 1.f / (fmaxf(d1, 0.f) + 1e-8f);
    float r2 = 1.f / (fmaxf(d2, 0.f) + 1e-8f);
    float inv = 1.f / (r0 + r1 + r2);
    size_t o = ((size_t)b * N1 + n) * 3;
    nidx[o] = i0; nidx[o + 1] = i1; nidx[o + 2] = i2;
    nw[o] = r0 * inv; nw[o + 1] = r1 * inv; nw[o + 2] = r2 * inv;
}

__global__ void k_interp(const float* __restrict__ p2, const int* __restrict__ nidx,
                         const float* __restrict__ nw, float* __restrict__ out,
                         int N1, int S2, int Pn, long long total)
{
    long long t = (long long)blockIdx.x * blockDim.x + threadIdx.x;
    if (t >= total) return;
    int c = (int)(t / Pn);
    int p = (int)(t - (long long)c * Pn);
    int b = p / N1;
    size_t o = (size_t)p * 3;
    const float* row = p2 + (size_t)c * (BBATCH * S2) + (size_t)b * S2;
    out[t] = nw[o] * row[nidx[o]] + nw[o + 1] * row[nidx[o + 1]] +
             nw[o + 2] * row[nidx[o + 2]];
}

// ---------------------------------------------------------------------------
// Head pooling: gfT[1280][32] = [max(l4,S4) ; max(l0u,N) ; mean(l0u,N)]^T
// ---------------------------------------------------------------------------
__global__ void k_pool(const float* __restrict__ l4, const float* __restrict__ l0u,
                       float* __restrict__ gfT)
{
    int t = blockIdx.x * blockDim.x + threadIdx.x;
    if (t >= 1280 * 32) return;
    int row = t >> 5, b = t & 31;
    float v;
    if (row < 1024) {
        const float* src = l4 + (size_t)row * (32 * 38) + b * 38;
        v = src[0];
        for (int s = 1; s < 38; ++s) v = fmaxf(v, src[s]);
    } else if (row < 1152) {
        const float* src = l0u + (size_t)(row - 1024) * (32 * NPTS) + b * NPTS;
        v = src[0];
        for (int n = 1; n < NPTS; ++n) v = fmaxf(v, src[n]);
    } else {
        const float* src = l0u + (size_t)(row - 1152) * (32 * NPTS) + b * NPTS;
        float s = 0.f;
        for (int n = 0; n < NPTS; ++n) s += src[n];
        v = s * (1.f / (float)NPTS);
    }
    gfT[t] = v;
}

__global__ void k_store_out(const float* __restrict__ ho, float* __restrict__ out)
{
    int t = blockIdx.x * blockDim.x + threadIdx.x;
    if (t >= 9600) return;
    int b = t / 300, m = t % 300;
    out[t] = ho[m * 32 + b];
}

// ---------------------------------------------------------------------------
// Host orchestration
// ---------------------------------------------------------------------------
struct CP { const float* w; const float* g; const float* b; int cout; };

static inline int imax(int a, int b) { return a > b ? a : b; }

extern "C" void kernel_launch(void* const* d_in, const int* in_sizes, int n_in,
                              void* d_out, int out_size, void* d_ws, size_t ws_size,
                              hipStream_t stream)
{
    (void)in_sizes; (void)n_in; (void)out_size; (void)ws_size;

    const float* xyz0 = (const float*)d_in[0];   // [B][N][3]
    int pi = 1;
    auto takeConv = [&](int cout) -> CP {
        CP c;
        c.w = (const float*)d_in[pi];
        c.g = (const float*)d_in[pi + 1];
        c.b = (const float*)d_in[pi + 2];
        c.cout = cout; pi += 3;
        return c;
    };
    CP im  = takeConv(64);
    CP s1a = takeConv(64),  s1b = takeConv(64),  s1c = takeConv(128);
    CP s2a = takeConv(128), s2b = takeConv(128), s2c = takeConv(256);
    CP s3a = takeConv(256), s3b = takeConv(256), s3c = takeConv(512);
    CP s4a = takeConv(512), s4b = takeConv(512), s4c = takeConv(1024);
    CP f4a = takeConv(512), f4b = takeConv(512);
    CP f3a = takeConv(512), f3b = takeConv(256);
    CP f2a = takeConv(256), f2b = takeConv(128);
    CP f1a = takeConv(128), f1b = takeConv(128), f1c = takeConv(128);
    const float* hw1  = (const float*)d_in[pi++];
    const float* hb1  = (const float*)d_in[pi++];
    const float* hg1  = (const float*)d_in[pi++];
    const float* hbb1 = (const float*)d_in[pi++];
    const float* hw2  = (const float*)d_in[pi++];
    const float* hb2  = (const float*)d_in[pi++];
    const float* hg2  = (const float*)d_in[pi++];
    const float* hbb2 = (const float*)d_in[pi++];
    const float* hwo  = (const float*)d_in[pi++];
    const float* hbo  = (const float*)d_in[pi++];

    // ---- workspace bump allocator ----
    char* wsbase = (char*)d_ws;
    size_t off = 0;
    auto alloc = [&](size_t bytes) -> void* {
        off = (off + 255) & ~(size_t)255;
        void* p = wsbase + off;
        off += bytes;
        return p;
    };

    const int B  = BBATCH;
    const int S1 = 612, S2n = 306, S3 = 76, S4 = 38;
    const int BN0 = B * NPTS;

    // persistent tensors
    float* l0   = (float*)alloc((size_t)64  * BN0 * 4);
    float* xyz1 = (float*)alloc((size_t)B * S1 * 3 * 4);
    float* l1   = (float*)alloc((size_t)128 * B * S1 * 4);
    float* xyz2 = (float*)alloc((size_t)B * S2n * 3 * 4);
    float* l2   = (float*)alloc((size_t)256 * B * S2n * 4);
    float* xyz3 = (float*)alloc((size_t)B * S3 * 3 * 4);
    float* l3   = (float*)alloc((size_t)512 * B * S3 * 4);
    float* xyz4 = (float*)alloc((size_t)B * S4 * 3 * 4);
    float* l4   = (float*)alloc((size_t)1024 * B * S4 * 4);
    float* l3u  = (float*)alloc((size_t)512 * B * S3 * 4);
    float* l2u  = (float*)alloc((size_t)256 * B * S2n * 4);
    float* l1u  = (float*)alloc((size_t)128 * B * S1 * 4);
    float* l0u  = (float*)alloc((size_t)128 * BN0 * 4);
    float* gfT  = (float*)alloc((size_t)1280 * 32 * 4);
    float* h1   = (float*)alloc((size_t)1024 * 32 * 4);
    float* h2   = (float*)alloc((size_t)512 * 32 * 4);
    float* ho   = (float*)alloc((size_t)300 * 32 * 4);
    float* stats = (float*)alloc((size_t)2 * 1024 * 4);
    const size_t scratchStart = off;

    // ---- helpers ----
    auto gemm = [&](const float* W, const float* X, float* Y, int M, int K, int P,
                    const float* bias, int relu) {
        dim3 gr((unsigned)((P + 127) / 128), (unsigned)((M + 63) / 64));
        k_gemm<<<gr, 256, 0, stream>>>(W, X, Y, M, K, P, bias, relu);
    };
    auto bnrun = [&](float* Y, int M, int P, const float* g, const float* bb, int relu) {
        k_bnstats<<<M, 256, 0, stream>>>(Y, P, stats);
        long long tot = (long long)M * P;
        k_bnapply<<<(unsigned)((tot + 255) / 256), 256, 0, stream>>>(
            Y, stats, g, bb, P, tot, relu);
    };

    auto run_sa = [&](const float* xyzPrev, const float* feats, int Nprev, int Cf,
                      int S, CP c0, CP c1, CP c2, float* xyzOut, float* featsOut) {
        off = scratchStart;
        int BS = B * S;
        long long Q = (long long)BS * KNN;
        int* fidx = (int*)alloc((size_t)BS * 4);
        int* kidx = (int*)alloc((size_t)Q * 4);
        k_fps<<<B, 256, 0, stream>>>(xyzPrev, Nprev, S, fidx);
        k_gather_xyz<<<(unsigned)((BS + 255) / 256), 256, 0, stream>>>(
            xyzPrev, fidx, xyzOut, Nprev, S, BS);
        k_knn<<<dim3((unsigned)((S + 127) / 128), (unsigned)B), 128, 0, stream>>>(
            xyzPrev, xyzOut, Nprev, S, kidx);
        float* buf0 = (float*)alloc((size_t)imax(Cf + 3, c1.cout) * Q * 4);
        float* buf1 = (float*)alloc((size_t)imax(c0.cout, c2.cout) * Q * 4);
        k_group<<<(unsigned)((Q + 255) / 256), 256, 0, stream>>>(
            xyzPrev, xyzOut, feats, kidx, buf0, Nprev, S, Cf, B * Nprev, Q);
        gemm(c0.w, buf0, buf1, c0.cout, Cf + 3, (int)Q, nullptr, 0);
        bnrun(buf1, c0.cout, (int)Q, c0.g, c0.b, 1);
        gemm(c1.w, buf1, buf0, c1.cout, c0.cout, (int)Q, nullptr, 0);
        bnrun(buf0, c1.cout, (int)Q, c1.g, c1.b, 1);
        gemm(c2.w, buf0, buf1, c2.cout, c1.cout, (int)Q, nullptr, 0);
        bnrun(buf1, c2.cout, (int)Q, c2.g, c2.b, 1);
        long long tot = (long long)c2.cout * BS;
        k_maxk<<<(unsigned)((tot + 255) / 256), 256, 0, stream>>>(
            buf1, featsOut, c2.cout, BS);
    };

    auto run_fp = [&](const float* xyzN, const float* xyzS,
                      const float* p1, int C1, const float* p2, int C2,
                      int N1, int S2, const CP* cv, int ncv, float* outFeats) {
        off = scratchStart;
        int Pn = B * N1;
        int*   nidx = (int*)alloc((size_t)Pn * 3 * 4);
        float* nw   = (float*)alloc((size_t)Pn * 3 * 4);
        k_3nn<<<dim3((unsigned)((N1 + 127) / 128), (unsigned)B), 128, 0, stream>>>(
            xyzN, xyzS, N1, S2, nidx, nw);
        float* Xc = (float*)alloc((size_t)(C1 + C2) * Pn * 4);
        hipMemcpyAsync(Xc, p1, (size_t)C1 * Pn * 4, hipMemcpyDeviceToDevice, stream);
        long long itot = (long long)C2 * Pn;
        k_interp<<<(unsigned)((itot + 255) / 256), 256, 0, stream>>>(
            p2, nidx, nw, Xc + (size_t)C1 * Pn, N1, S2, Pn, itot);
        int maxc = 0;
        for (int i = 0; i < ncv; ++i) maxc = imax(maxc, cv[i].cout);
        float* bufA = (float*)alloc((size_t)maxc * Pn * 4);
        float* bufB = (float*)alloc((size_t)maxc * Pn * 4);
        const float* cur = Xc;
        int cin = C1 + C2;
        for (int i = 0; i < ncv; ++i) {
            float* dst = (i == ncv - 1) ? outFeats : ((i & 1) ? bufB : bufA);
            gemm(cv[i].w, cur, dst, cv[i].cout, cin, Pn, nullptr, 0);
            bnrun(dst, cv[i].cout, Pn, cv[i].g, cv[i].b, 1);
            cur = dst; cin = cv[i].cout;
        }
    };

    // ---- pipeline ----
    // input MLP: [3][B*N] -> l0 [64][B*N]
    off = scratchStart;
    float* X0 = (float*)alloc((size_t)3 * BN0 * 4);
    k_xyz_cmaj<<<(unsigned)((BN0 + 255) / 256), 256, 0, stream>>>(xyz0, X0, BN0);
    gemm(im.w, X0, l0, 64, 3, BN0, nullptr, 0);
    bnrun(l0, 64, BN0, im.g, im.b, 1);

    run_sa(xyz0, l0, NPTS, 64,  S1,  s1a, s1b, s1c, xyz1, l1);
    run_sa(xyz1, l1, S1,   128, S2n, s2a, s2b, s2c, xyz2, l2);
    run_sa(xyz2, l2, S2n,  256, S3,  s3a, s3b, s3c, xyz3, l3);
    run_sa(xyz3, l3, S3,   512, S4,  s4a, s4b, s4c, xyz4, l4);

    CP fp4c[2] = { f4a, f4b };
    run_fp(xyz3, xyz4, l3, 512, l4,  1024, S3,  S4,  fp4c, 2, l3u);
    CP fp3c[2] = { f3a, f3b };
    run_fp(xyz2, xyz3, l2, 256, l3u, 512,  S2n, S3,  fp3c, 2, l2u);
    CP fp2c[2] = { f2a, f2b };
    run_fp(xyz1, xyz2, l1, 128, l2u, 256,  S1,  S2n, fp2c, 2, l1u);
    CP fp1c[3] = { f1a, f1b, f1c };
    run_fp(xyz0, xyz1, l0, 64,  l1u, 128,  NPTS, S1, fp1c, 3, l0u);

    // head
    off = scratchStart;
    k_pool<<<(unsigned)((1280 * 32 + 255) / 256), 256, 0, stream>>>(l4, l0u, gfT);
    gemm(hw1, gfT, h1, 1024, 1280, 32, hb1, 1);
    bnrun(h1, 1024, 32, hg1, hbb1, 0);
    gemm(hw2, h1, h2, 512, 1024, 32, hb2, 1);
    bnrun(h2, 512, 32, hg2, hbb2, 0);
    gemm(hwo, h2, ho, 300, 512, 32, hbo, 0);
    k_store_out<<<(unsigned)((9600 + 255) / 256), 256, 0, stream>>>(ho, (float*)d_out);
}
